// GumbelTernaryLinear_15058155339889
// MI455X (gfx1250) — matrix operand order repaired and checked by reading, not compiled, not run on hardware
//
#include <hip/hip_runtime.h>
#include <hip/hip_bf16.h>

// Problem dims (match reference)
#define OUT_F 4096
#define IN_F  4096
#define TOK   1024
#define GRP   128

typedef __attribute__((ext_vector_type(16))) _Float16 v16h;
typedef __attribute__((ext_vector_type(8)))  _Float16 v8h;
typedef __attribute__((ext_vector_type(8)))  float    v8f;

// ---------------------------------------------------------------------------
// Kernel 1: fuse sharp-softmax ternarization + group scale + mask -> f16 W.
// Pure streaming: 201 MB logits + 64 MB mask in, 32 MB f16 W out.
// ---------------------------------------------------------------------------
__global__ void __launch_bounds__(256)
build_w_kernel(const float* __restrict__ logits,
               const float* __restrict__ scales,
               const int*   __restrict__ mask,
               _Float16*    __restrict__ wq) {
  const size_t idx = (size_t)blockIdx.x * blockDim.x + threadIdx.x; // over OUT*IN
  const float l0 = logits[3 * idx + 0];
  const float l1 = logits[3 * idx + 1];
  const float l2 = logits[3 * idx + 2];
  // softmax(l / 0.01) -> e2 - e0 (max-subtracted for stability)
  const float m  = fmaxf(l0, fmaxf(l1, l2));
  const float e0 = __expf((l0 - m) * 100.0f);
  const float e1 = __expf((l1 - m) * 100.0f);
  const float e2 = __expf((l2 - m) * 100.0f);
  float w = (e2 - e0) / (e0 + e1 + e2);
  const int o = (int)(idx >> 12);     // idx / 4096
  const int i = (int)(idx & 4095);    // idx % 4096
  w *= scales[o * (IN_F / GRP) + (i / GRP)];
  w *= (float)mask[idx];
  wq[idx] = (_Float16)w;
}

// ---------------------------------------------------------------------------
// Kernel 2: x (f32) -> f16 activations for the WMMA path.
// ---------------------------------------------------------------------------
__global__ void __launch_bounds__(256)
convert_x_kernel(const float* __restrict__ x, _Float16* __restrict__ xh) {
  const size_t idx = (size_t)blockIdx.x * blockDim.x + threadIdx.x; // over TOK*IN
  xh[idx] = (_Float16)x[idx];
}

// ---------------------------------------------------------------------------
// Kernel 3: out[TOK, OUT] = Xh[TOK, IN] * Wq[OUT, IN]^T via v_wmma_f32_16x16x32_f16.
// Block = 256 threads = 8 waves arranged 2 (M) x 4 (N).
// Each wave computes a 64x16 C panel (4 stacked 16x16 accumulators) so each
// B fragment is reused 4x from registers. Block tile = 128 (M) x 64 (N).
// W(32MB f16) + X(8MB f16) are L2-resident (192MB L2), so the K-loop streams
// from L2, not HBM.
// ---------------------------------------------------------------------------
__global__ void __launch_bounds__(256)
wmma_gemm_kernel(const _Float16* __restrict__ Xh,
                 const _Float16* __restrict__ Wq,
                 float* __restrict__ out) {
  const int lane    = threadIdx.x & 31;
  const int wave    = threadIdx.x >> 5;
  const int waveN   = wave & 3;   // 0..3 -> N direction
  const int waveM   = wave >> 2;  // 0..1 -> M direction
  const int halfSel = lane >> 4;  // 0 or 1 (lane half)
  const int l16     = lane & 15;

  // B: lane holds column N = l16 of the 32x16 fragment, K-range halfSel*16..+15
  const int oCol  = blockIdx.x * 64 + waveN * 16 + l16;
  // A: lane holds row M = l16; K-range halfSel*16..+15
  const int tRow0 = blockIdx.y * 128 + waveM * 64;

  const _Float16* __restrict__ wrow = Wq + (size_t)oCol * IN_F + halfSel * 16;
  const _Float16* __restrict__ xrow[4];
#pragma unroll
  for (int s = 0; s < 4; ++s)
    xrow[s] = Xh + (size_t)(tRow0 + s * 16 + l16) * IN_F + halfSel * 16;

  v8f acc[4] = {};

  for (int k = 0; k < IN_F; k += 32) {
    // B fragment: 16 contiguous halfs (32 B)
    const v16h bfrag = *(const v16h*)(wrow + k);
#pragma unroll
    for (int s = 0; s < 4; ++s) {
      // A fragment: two contiguous 8-half (16 B) chunks (k = 16*halfSel + s, same order as B)
      const v8h alo = *(const v8h*)(xrow[s] + k);
      const v8h ahi = *(const v8h*)(xrow[s] + k + 8);
      const v16h afrag = __builtin_shufflevector(
          alo, ahi, 0, 1, 2, 3, 4, 5, 6, 7, 8, 9, 10, 11, 12, 13, 14, 15);
      // D = A x B + C  (f32 accumulate)
      acc[s] = __builtin_amdgcn_wmma_f32_16x16x32_f16(
          /*neg_a=*/false, afrag, /*neg_b=*/false, bfrag,
          /*c_mod=*/(short)0, acc[s], /*reuse_a=*/false, /*reuse_b=*/false);
    }
  }

  // C/D layout: VGPR r, lane -> M = r + 8*halfSel, N = l16
#pragma unroll
  for (int s = 0; s < 4; ++s) {
    const int mBase = tRow0 + s * 16 + 8 * halfSel;
    float* __restrict__ op = out + (size_t)mBase * OUT_F + oCol;
#pragma unroll
    for (int r = 0; r < 8; ++r)
      op[(size_t)r * OUT_F] = acc[s][r];
  }
}

// ---------------------------------------------------------------------------
extern "C" void kernel_launch(void* const* d_in, const int* in_sizes, int n_in,
                              void* d_out, int out_size, void* d_ws, size_t ws_size,
                              hipStream_t stream) {
  // setup_inputs() order: x, logits, scales, mask
  const float* x      = (const float*)d_in[0];
  const float* logits = (const float*)d_in[1];
  const float* scales = (const float*)d_in[2];
  const int*   mask   = (const int*)d_in[3];
  float* out = (float*)d_out;

  // Workspace: f16 W [OUT,IN] (32 MB) then f16 X [TOK,IN] (8 MB)
  _Float16* Wq = (_Float16*)d_ws;
  _Float16* Xh = (_Float16*)((char*)d_ws + (size_t)OUT_F * IN_F * sizeof(_Float16));

  build_w_kernel<<<(OUT_F * IN_F) / 256, 256, 0, stream>>>(logits, scales, mask, Wq);
  convert_x_kernel<<<(TOK * IN_F) / 256, 256, 0, stream>>>(x, Xh);

  dim3 grid(OUT_F / 64, TOK / 128);  // 64 x 8 blocks, 8 waves each
  wmma_gemm_kernel<<<grid, 256, 0, stream>>>(Xh, Wq, out);
}
